// FocusAttention_63496796504511
// MI455X (gfx1250) — matrix operand order repaired and checked by reading, not compiled, not run on hardware
//
#include <hip/hip_runtime.h>
#include <hip/hip_bf16.h>

typedef __attribute__((ext_vector_type(16))) _Float16 v16h;
typedef __attribute__((ext_vector_type(8)))  _Float16 v8h;
typedef __attribute__((ext_vector_type(8)))  float    v8f;

#define Cc   32
#define Nn   4096
#define Bb   8
#define POOL 1024

// ---------------- K1: 1x1 conv 64->32 on 32x32 ----------------
__global__ __launch_bounds__(256) void k_tlconv(const float* __restrict__ xin,
    const float* __restrict__ tlw, const float* __restrict__ tlb, float* __restrict__ xs) {
  int idx = blockIdx.x * 256 + threadIdx.x;              // (b*32+c)*1024+p
  if (idx >= Bb * 32 * 1024) return;
  int p = idx & 1023, bc = idx >> 10, c = bc & 31, b = bc >> 5;
  float acc = tlb[c];
  const float* wr = tlw + c * 64;
  const float* xi = xin + ((size_t)b * 64) * 1024 + p;
  #pragma unroll 8
  for (int ci = 0; ci < 64; ++ci) acc += wr[ci] * xi[ci * 1024];
  xs[idx] = acc;
}

// ---------------- K2: bilinear x2 upsample (half-pixel) ----------------
__global__ __launch_bounds__(256) void k_upsample(const float* __restrict__ xs,
    float* __restrict__ xr, float* __restrict__ dout0) {
  int idx = blockIdx.x * 256 + threadIdx.x;              // (b*32+c)*4096+n
  if (idx >= Bb * Cc * Nn) return;
  int n = idx & 4095, bc = idx >> 12;
  int h = n >> 6, w = n & 63;
  float sy = (h + 0.5f) * 0.5f - 0.5f, sx = (w + 0.5f) * 0.5f - 0.5f;
  float y0f = floorf(sy), x0f = floorf(sx);
  float fy = sy - y0f, fx = sx - x0f;
  int y0 = (int)y0f, x0 = (int)x0f;
  int y0c = min(max(y0, 0), 31), y1c = min(max(y0 + 1, 0), 31);
  int x0c = min(max(x0, 0), 31), x1c = min(max(x0 + 1, 0), 31);
  const float* src = xs + (size_t)bc * 1024;
  float v = (1.f - fy) * ((1.f - fx) * src[y0c * 32 + x0c] + fx * src[y0c * 32 + x1c])
          +        fy  * ((1.f - fx) * src[y1c * 32 + x0c] + fx * src[y1c * 32 + x1c]);
  xr[idx] = v;
  dout0[idx] = v;
}

// ---------------- K3: kv = f @ kv_w^T, k l2norm; write kv_img (B,64,H,W) ----------------
__global__ __launch_bounds__(256) void k_kv(const float* __restrict__ f,
    const float* __restrict__ kvw, float* __restrict__ kvimg) {
  int idx = blockIdx.x * 256 + threadIdx.x;              // b*N+n
  if (idx >= Bb * Nn) return;
  int b = idx >> 12, n = idx & 4095;
  float fr[32];
  #pragma unroll
  for (int c = 0; c < 32; ++c) fr[c] = f[(((size_t)b * 32 + c) << 12) + n];
  float kv[64];
  for (int o = 0; o < 64; ++o) {
    float a = 0.f; const float* wr = kvw + o * 32;
    #pragma unroll
    for (int c = 0; c < 32; ++c) a += wr[c] * fr[c];
    kv[o] = a;
  }
  float nrm = 0.f;
  #pragma unroll
  for (int c = 0; c < 32; ++c) nrm += kv[c] * kv[c];
  float ik = 1.f / fmaxf(sqrtf(nrm), 1e-12f);
  #pragma unroll
  for (int c = 0; c < 32; ++c) {
    kvimg[(((size_t)b * 64 + c) << 12) + n] = kv[c] * ik;
    kvimg[(((size_t)b * 64 + 32 + c) << 12) + n] = kv[32 + c];
  }
}

// ---------------- K4: q, q_norm, q_scaled (f32 + WMMA-B f16), tokq ----------------
__global__ __launch_bounds__(256) void k_q(const float* __restrict__ xr,
    const float* __restrict__ qw, const float* __restrict__ qb,
    const float* __restrict__ qemb, const float* __restrict__ temp,
    const float* __restrict__ tokens, const float* __restrict__ lbias,
    float* __restrict__ qsc, _Float16* __restrict__ qB, float* __restrict__ tokq) {
  int idx = blockIdx.x * 256 + threadIdx.x;              // b*N+n
  if (idx >= Bb * Nn) return;
  int b = idx >> 12, n = idx & 4095;
  int h = n >> 6, w = n & 63;
  float xrow[32];
  #pragma unroll
  for (int c = 0; c < 32; ++c) xrow[c] = xr[(((size_t)b * 32 + c) << 12) + n];
  float q[32];
  for (int c = 0; c < 32; ++c) {
    float a = qb[c]; const float* wr = qw + c * 32;
    #pragma unroll
    for (int j = 0; j < 32; ++j) a += wr[j] * xrow[j];
    q[c] = a;
  }
  float nrm = 0.f;
  #pragma unroll
  for (int c = 0; c < 32; ++c) nrm += q[c] * q[c];
  float ik = 1.f / fmaxf(sqrtf(nrm), 1e-12f);
  float qn[32];
  #pragma unroll
  for (int c = 0; c < 32; ++c) qn[c] = q[c] * ik;
  float sp = log1pf(__expf(temp[0]));
  int cnt = (3 - (h == 0) - (h == 63)) * (3 - (w == 0) - (w == 63));
  float ss = logf((float)cnt + 4096.0f);
  int tile = (b << 8) + (n >> 4);
  // B-matrix 16-bit layout: lane = n + 16*((K>>3)&1), half j = (K&7) + 8*(K>>4)
  #pragma unroll
  for (int c = 0; c < 32; ++c) {
    float qs = (qn[c] + qemb[c]) * sp * ss;
    qsc[(size_t)idx * 32 + c] = qs;
    qB[(((size_t)tile * 32) + (n & 15) + 16 * ((c >> 3) & 1)) * 16 + (c & 7) + ((c >> 4) << 3)] = (_Float16)qs;
  }
  #pragma unroll
  for (int j = 0; j < 9; ++j) {
    float a = lbias[j];
    #pragma unroll
    for (int c = 0; c < 32; ++c) a += qn[c] * tokens[c * 9 + j];
    tokq[(size_t)idx * 9 + j] = a;
  }
}

// ---------------- K5: local 3x3 logits (zero-padded unfold => OOB logit = 0) ----------------
__global__ __launch_bounds__(256) void k_lloc(const float* __restrict__ qsc,
    const float* __restrict__ kvimg, float* __restrict__ lloc) {
  int idx = blockIdx.x * 256 + threadIdx.x;
  if (idx >= Bb * Nn) return;
  int b = idx >> 12, n = idx & 4095;
  int h = n >> 6, w = n & 63;
  float qs[32];
  #pragma unroll
  for (int c = 0; c < 32; ++c) qs[c] = qsc[(size_t)idx * 32 + c];
  const float* kimg = kvimg + ((size_t)b * 64 << 12);
  int j = 0;
  for (int dh = -1; dh <= 1; ++dh)
    for (int dw = -1; dw <= 1; ++dw, ++j) {
      int hh = h + dh, ww = w + dw;
      float a = 0.f;
      if (hh >= 0 && hh < 64 && ww >= 0 && ww < 64) {
        const float* kp = kimg + hh * 64 + ww;
        #pragma unroll
        for (int c = 0; c < 32; ++c) a += qs[c] * kp[(size_t)c << 12];
      }
      lloc[(size_t)idx * 9 + j] = a;
    }
}

// ---------------- K6: sr-conv + gelu + 2x2 pool + LN + kvp -> WMMA f16 tiles ----------------
__global__ __launch_bounds__(256) void k_pool(const float* __restrict__ f,
    const float* __restrict__ srw, const float* __restrict__ srb,
    const float* __restrict__ lng, const float* __restrict__ lnb,
    const float* __restrict__ kvw,
    _Float16* __restrict__ kA, _Float16* __restrict__ vB) {
  int idx = blockIdx.x * 256 + threadIdx.x;              // b*1024+p
  if (idx >= Bb * POOL) return;
  int b = idx >> 10, p = idx & 1023;
  int ph = p >> 5, pw = p & 31;
  float t[32];
  #pragma unroll
  for (int c = 0; c < 32; ++c) t[c] = 0.f;
  for (int py = 0; py < 2; ++py)
    for (int px = 0; px < 2; ++px) {
      int n = (ph * 2 + py) * 64 + (pw * 2 + px);
      float fp[32];
      #pragma unroll
      for (int ci = 0; ci < 32; ++ci) fp[ci] = f[(((size_t)b * 32 + ci) << 12) + n];
      for (int c = 0; c < 32; ++c) {
        float s = srb[c]; const float* wr = srw + c * 32;
        #pragma unroll
        for (int ci = 0; ci < 32; ++ci) s += wr[ci] * fp[ci];
        t[c] += 0.5f * s * (1.0f + erff(s * 0.70710678118654752f));
      }
    }
  float mean = 0.f;
  #pragma unroll
  for (int c = 0; c < 32; ++c) { t[c] *= 0.25f; mean += t[c]; }
  mean *= (1.f / 32.f);
  float var = 0.f;
  #pragma unroll
  for (int c = 0; c < 32; ++c) { float d = t[c] - mean; var += d * d; }
  var *= (1.f / 32.f);
  float inv = rsqrtf(var + 1e-5f);
  float tn[32];
  #pragma unroll
  for (int c = 0; c < 32; ++c) tn[c] = (t[c] - mean) * inv * lng[c] + lnb[c];
  float kv[64];
  for (int o = 0; o < 64; ++o) {
    float a = 0.f; const float* wr = kvw + o * 32;
    #pragma unroll
    for (int c = 0; c < 32; ++c) a += wr[c] * tn[c];
    kv[o] = a;
  }
  float nrm = 0.f;
  #pragma unroll
  for (int c = 0; c < 32; ++c) nrm += kv[c] * kv[c];
  float ik = 1.f / fmaxf(sqrtf(nrm), 1e-12f);
  int tile = p >> 4, kk = p & 31, chunk = p >> 5;
  #pragma unroll
  for (int c = 0; c < 32; ++c) {
    // K into A-matrix 16-bit layout: lane = row + 16*sel, K groups {kb..kb+7, kb+16..kb+23}
    int sel = (c >> 3) & 1;
    int jj = (c & 7) + ((c >> 4) << 3);
    size_t ki = (((size_t)(b * 64 + tile)) * 32 + (p & 15) + 16 * sel) * 16 + jj;
    kA[ki] = (_Float16)(kv[c] * ik);
    size_t vi = ((((size_t)(b * 32 + chunk)) * 2 + (c >> 4)) * 32 + (c & 15) + 16 * (kk >> 4)) * 16 + (kk & 15);
    vB[vi] = (_Float16)kv[32 + c];
  }
}

// ---------------- K7: flash pooled attention, transposed scores (query = lane dim) ----------------
__global__ __launch_bounds__(256) void k_flash(const _Float16* __restrict__ qBh,
    const _Float16* __restrict__ kAh, const _Float16* __restrict__ vBh,
    float* __restrict__ spool, float* __restrict__ mrow, float* __restrict__ zrow) {
  __shared__ _Float16 P[8][16 * 32];
  const v16h* qB = (const v16h*)qBh;
  const v16h* kA = (const v16h*)kAh;
  const v16h* vB = (const v16h*)vBh;
  int wave = threadIdx.x >> 5, lane = threadIdx.x & 31;
  int qt = blockIdx.x * 8 + wave;                        // 0..2047
  int b = qt >> 8, tIn = qt & 255;
  int hi = lane >> 4, col = lane & 15;
  v16h bq = qB[(size_t)qt * 32 + lane];                  // held in regs across loops
  const v16h* kbase = kA + (size_t)b * 64 * 32;
  const v16h* vbase = vB + (size_t)b * 32 * 2 * 32;

  // pass 1: S^T = K x Q^T; row-max per query is an in-register reduction over 8 accum regs
  float mlane = -3.0e38f;
  for (int t = 0; t < 64; ++t) {
    v8f z = {};
    v8f d = __builtin_amdgcn_wmma_f32_16x16x32_f16(false, kbase[t * 32 + lane], false,
                                                   bq, (short)0, z, false, false);
    if (t + 1 < 64) __builtin_prefetch(&kbase[(t + 1) * 32 + lane], 0, 0);
    #pragma unroll
    for (int r = 0; r < 8; ++r) mlane = fmaxf(mlane, d[r]);
  }
  // merge key-halves: lane l and l^16 cover complementary keys of the same query
  float mq = fmaxf(mlane, __shfl_xor(mlane, 16, 32));

  // pass 2: P = exp(s - mq) in f16; acc += P @ V ; Z via ones-column WMMA
  v8f acc0 = {}, acc1 = {}, accZ = {};
  v16h ones;
  #pragma unroll
  for (int j = 0; j < 16; ++j) ones[j] = (_Float16)1.0f;
  _Float16* Pw = P[wave];
  for (int ch = 0; ch < 32; ++ch) {
    v8f z = {};
    v8f d0 = __builtin_amdgcn_wmma_f32_16x16x32_f16(false, kbase[(2 * ch) * 32 + lane], false,
                                                    bq, (short)0, z, false, false);
    v8f d1 = __builtin_amdgcn_wmma_f32_16x16x32_f16(false, kbase[(2 * ch + 1) * 32 + lane], false,
                                                    bq, (short)0, z, false, false);
    v8h e0, e1;
    #pragma unroll
    for (int r = 0; r < 8; ++r) e0[r] = (_Float16)__expf(d0[r] - mq);
    #pragma unroll
    for (int r = 0; r < 8; ++r) e1[r] = (_Float16)__expf(d1[r] - mq);
    // P[query=col][key]: lane's 8 exp values are contiguous keys -> 16B LDS stores
    *(v8h*)(Pw + col * 32 + hi * 8)      = e0;   // keys hi*8 .. hi*8+7
    *(v8h*)(Pw + col * 32 + 16 + hi * 8) = e1;   // keys 16+hi*8 ..
    // re-read P as A-matrix (rows = queries)
    const v8h* pr = (const v8h*)(Pw + col * 32);
    v8h lo  = pr[2 * hi];       // K = 16*hi..16*hi+7
    v8h hiv = pr[2 * hi + 1];   // K = 16*hi+8..16*hi+15
    v16h ap;
    #pragma unroll
    for (int j = 0; j < 8; ++j) { ap[j] = lo[j]; ap[8 + j] = hiv[j]; }
    acc0 = __builtin_amdgcn_wmma_f32_16x16x32_f16(false, ap, false,
             vbase[(ch * 2 + 0) * 32 + lane], (short)0, acc0, false, false);
    acc1 = __builtin_amdgcn_wmma_f32_16x16x32_f16(false, ap, false,
             vbase[(ch * 2 + 1) * 32 + lane], (short)0, acc1, false, false);
    accZ = __builtin_amdgcn_wmma_f32_16x16x32_f16(false, ap, false,
             ones, (short)0, accZ, false, false);
  }

  int qbase = b * Nn + tIn * 16;
  #pragma unroll
  for (int r = 0; r < 8; ++r) {
    int m = r + 8 * hi;                                  // query row
    int q = qbase + m;
    spool[(size_t)q * 32 + col]      = acc0[r];
    spool[(size_t)q * 32 + 16 + col] = acc1[r];
    if (col == 0) zrow[q] = accZ[r];
  }
  if (hi == 0) mrow[qbase + col] = mq;
}

// ---------------- K8: epilogue — merge local branch, softmax, x_local, proj ----------------
__global__ __launch_bounds__(256) void k_epi(const float* __restrict__ spool,
    const float* __restrict__ mrow, const float* __restrict__ zrow,
    const float* __restrict__ lloc, const float* __restrict__ tokq,
    const float* __restrict__ kvimg,
    const float* __restrict__ projw, const float* __restrict__ projb,
    float* __restrict__ ctx) {
  int idx = blockIdx.x * 256 + threadIdx.x;
  if (idx >= Bb * Nn) return;
  int b = idx >> 12, n = idx & 4095;
  int h = n >> 6, w = n & 63;
  float mp = mrow[idx], Z0 = zrow[idx];
  float ll[9], mx = mp;
  #pragma unroll
  for (int j = 0; j < 9; ++j) { ll[j] = lloc[(size_t)idx * 9 + j]; mx = fmaxf(mx, ll[j]); }
  float scale = __expf(mp - mx);
  float Z = Z0 * scale;
  float tk[9];
  #pragma unroll
  for (int j = 0; j < 9; ++j) { float e = __expf(ll[j] - mx); Z += e; tk[j] = e; }
  float invZ = 1.f / Z;
  #pragma unroll
  for (int j = 0; j < 9; ++j) tk[j] = tokq[(size_t)idx * 9 + j] + tk[j] * invZ;
  float cpre[32];
  float s2 = scale * invZ;
  #pragma unroll
  for (int c = 0; c < 32; ++c) cpre[c] = spool[(size_t)idx * 32 + c] * s2;
  const float* vimg = kvimg + (((size_t)b * 64 + 32) << 12);
  int j = 0;
  for (int dh = -1; dh <= 1; ++dh)
    for (int dw = -1; dw <= 1; ++dw, ++j) {
      int hh = h + dh, ww = w + dw;
      if (hh >= 0 && hh < 64 && ww >= 0 && ww < 64) {
        const float* vp = vimg + hh * 64 + ww;
        float tj = tk[j];
        #pragma unroll
        for (int c = 0; c < 32; ++c) cpre[c] += tj * vp[(size_t)c << 12];
      }
    }
  for (int co = 0; co < 32; ++co) {
    float a = projb[co]; const float* wr = projw + co * 32;
    #pragma unroll
    for (int c = 0; c < 32; ++c) a += wr[c] * cpre[c];
    ctx[(((size_t)b * 32 + co) << 12) + n] = a;
  }
}

// ---------------- K9: per-(b,c) spatial avg & max ----------------
__global__ __launch_bounds__(256) void k_castat(const float* __restrict__ f,
    float* __restrict__ avg, float* __restrict__ mx) {
  __shared__ float ss[256], sm[256];
  int bc = blockIdx.x;
  const float* p = f + ((size_t)bc << 12);
  float s = 0.f, m = -3.0e38f;
  for (int i = threadIdx.x; i < 4096; i += 256) { float v = p[i]; s += v; m = fmaxf(m, v); }
  ss[threadIdx.x] = s; sm[threadIdx.x] = m;
  __syncthreads();
  for (int off = 128; off > 0; off >>= 1) {
    if (threadIdx.x < off) {
      ss[threadIdx.x] += ss[threadIdx.x + off];
      sm[threadIdx.x] = fmaxf(sm[threadIdx.x], sm[threadIdx.x + off]);
    }
    __syncthreads();
  }
  if (threadIdx.x == 0) { avg[bc] = ss[0] * (1.f / 4096.f); mx[bc] = sm[0]; }
}

// ---------------- K10: channel-attention MLP (32->2->32), sigmoid ----------------
__global__ void k_camlp(const float* __restrict__ avg, const float* __restrict__ mx,
    const float* __restrict__ fc1, const float* __restrict__ fc2, float* __restrict__ caf) {
  int b = threadIdx.x;
  if (b >= Bb) return;
  const float* av = avg + b * 32;
  const float* mv = mx + b * 32;
  float ha[2], hm[2];
  for (int hh = 0; hh < 2; ++hh) {
    float a = 0.f, m2 = 0.f;
    #pragma unroll
    for (int c = 0; c < 32; ++c) { a += fc1[hh * 32 + c] * av[c]; m2 += fc1[hh * 32 + c] * mv[c]; }
    ha[hh] = fmaxf(a, 0.f); hm[hh] = fmaxf(m2, 0.f);
  }
  for (int c = 0; c < 32; ++c) {
    float o = fc2[c * 2 + 0] * ha[0] + fc2[c * 2 + 1] * ha[1]
            + fc2[c * 2 + 0] * hm[0] + fc2[c * 2 + 1] * hm[1];
    caf[b * 32 + c] = 1.f / (1.f + __expf(-o));
  }
}

// ---------------- K11: res = ctx * (ca*f) * xr ----------------
__global__ __launch_bounds__(256) void k_mul(const float* __restrict__ ctx,
    const float* __restrict__ caf, const float* __restrict__ f,
    const float* __restrict__ xr, float* __restrict__ res0) {
  int idx = blockIdx.x * 256 + threadIdx.x;
  if (idx >= Bb * Cc * Nn) return;
  int bc = idx >> 12;
  res0[idx] = ctx[idx] * (caf[bc] * f[idx]) * xr[idx];
}

// ---------------- K12: 3x3 conv C->C + ReLU, LDS tiled 16x16 ----------------
__global__ __launch_bounds__(256) void k_conv3(const float* __restrict__ in,
    const float* __restrict__ w9, const float* __restrict__ bias, float* __restrict__ out) {
  __shared__ float tile[32][18][18];
  int blk = blockIdx.x;                                  // b*16 + tile
  int b = blk >> 4, t = blk & 15;
  int ty0 = (t >> 2) * 16, tx0 = (t & 3) * 16;
  for (int i = threadIdx.x; i < 32 * 18 * 18; i += 256) {
    int ci = i / 324, rem = i % 324, yy = rem / 18, xx = rem % 18;
    int gy = ty0 + yy - 1, gx = tx0 + xx - 1;
    float v = 0.f;
    if (gy >= 0 && gy < 64 && gx >= 0 && gx < 64)
      v = in[(((size_t)b * 32 + ci) << 12) + gy * 64 + gx];
    tile[ci][yy][xx] = v;
  }
  __syncthreads();
  int ty = threadIdx.x >> 4, tx = threadIdx.x & 15;
  for (int co = 0; co < 32; ++co) {
    float acc = bias[co];
    const float* wp = w9 + co * 288;
    for (int ci = 0; ci < 32; ++ci) {
      const float* wr = wp + ci * 9;
      const float* tp = &tile[ci][ty][tx];
      acc += wr[0] * tp[0]  + wr[1] * tp[1]  + wr[2] * tp[2]
           + wr[3] * tp[18] + wr[4] * tp[19] + wr[5] * tp[20]
           + wr[6] * tp[36] + wr[7] * tp[37] + wr[8] * tp[38];
    }
    out[(((size_t)b * 32 + co) << 12) + (ty0 + ty) * 64 + (tx0 + tx)] = fmaxf(acc, 0.f);
  }
}

// ---------------- K13: final 1x1 conv C->1 + seg_map add ----------------
__global__ __launch_bounds__(256) void k_final(const float* __restrict__ res,
    const float* __restrict__ w, const float* __restrict__ bi,
    const float* __restrict__ seg, float* __restrict__ outp, float* __restrict__ resout) {
  int idx = blockIdx.x * 256 + threadIdx.x;              // b*N+n
  if (idx >= Bb * Nn) return;
  int b = idx >> 12, n = idx & 4095;
  float acc = bi[0];
  const float* rp = res + (((size_t)b * 32) << 12) + n;
  #pragma unroll
  for (int c = 0; c < 32; ++c) acc += w[c] * rp[(size_t)c << 12];
  resout[idx] = acc;
  outp[idx] = acc + seg[idx];
}

extern "C" void kernel_launch(void* const* d_in, const int* in_sizes, int n_in,
                              void* d_out, int out_size, void* d_ws, size_t ws_size,
                              hipStream_t stream) {
  const float* f    = (const float*)d_in[0];
  const float* xin  = (const float*)d_in[1];
  const float* seg  = (const float*)d_in[2];
  const float* tlw  = (const float*)d_in[3];
  const float* tlb  = (const float*)d_in[4];
  const float* qw   = (const float*)d_in[5];
  const float* qb   = (const float*)d_in[6];
  const float* qemb = (const float*)d_in[7];
  const float* temp = (const float*)d_in[8];
  const float* kvw  = (const float*)d_in[9];
  const float* srw  = (const float*)d_in[10];
  const float* srb  = (const float*)d_in[11];
  const float* lng  = (const float*)d_in[12];
  const float* lnb  = (const float*)d_in[13];
  const float* tok  = (const float*)d_in[14];
  const float* lbi  = (const float*)d_in[15];
  const float* pjw  = (const float*)d_in[16];
  const float* pjb  = (const float*)d_in[17];
  const float* fc1  = (const float*)d_in[18];
  const float* fc2  = (const float*)d_in[19];
  const float* c1w  = (const float*)d_in[20];
  const float* c1b  = (const float*)d_in[21];
  const float* c2w  = (const float*)d_in[22];
  const float* c2b  = (const float*)d_in[23];
  const float* c3w  = (const float*)d_in[24];
  const float* c3b  = (const float*)d_in[25];

  float* W = (float*)d_ws;
  float* xs    = W;               // 262144
  float* xr    = xs    + 262144;  // 1048576
  float* qsc   = xr    + 1048576; // 1048576
  float* tokq  = qsc   + 1048576; // 294912
  float* lloc  = tokq  + 294912;  // 294912
  float* kvimg = lloc  + 294912;  // 2097152
  float* spool = kvimg + 2097152; // 1048576
  float* mrow  = spool + 1048576; // 32768
  float* zrow  = mrow  + 32768;   // 32768
  float* ctx   = zrow  + 32768;   // 1048576
  float* res0  = ctx   + 1048576; // 1048576
  float* res1  = res0  + 1048576; // 1048576
  float* avgb  = res1  + 1048576; // 256
  float* mxb   = avgb  + 256;     // 256
  float* caf   = mxb   + 256;     // 256
  _Float16* qB = (_Float16*)(caf + 256);  // 1048576 halves
  _Float16* kA = qB + 1048576;            // 262144 halves
  _Float16* vB = kA + 262144;             // 262144 halves

  float* out0 = (float*)d_out;            // x_r  (B,C,H,W)
  float* out1 = out0 + Bb * Cc * Nn;      // out  (B,1,H,W)
  float* out2 = out1 + Bb * Nn;           // res_out

  k_tlconv  <<<1024, 256, 0, stream>>>(xin, tlw, tlb, xs);
  k_upsample<<<4096, 256, 0, stream>>>(xs, xr, out0);
  k_kv      <<<128,  256, 0, stream>>>(f, kvw, kvimg);
  k_q       <<<128,  256, 0, stream>>>(xr, qw, qb, qemb, temp, tok, lbi, qsc, qB, tokq);
  k_lloc    <<<128,  256, 0, stream>>>(qsc, kvimg, lloc);
  k_pool    <<<32,   256, 0, stream>>>(f, srw, srb, lng, lnb, kvw, kA, vB);
  k_flash   <<<256,  256, 0, stream>>>(qB, kA, vB, spool, mrow, zrow);
  k_epi     <<<128,  256, 0, stream>>>(spool, mrow, zrow, lloc, tokq, kvimg, pjw, pjb, ctx);
  k_castat  <<<256,  256, 0, stream>>>(f, avgb, mxb);
  k_camlp   <<<1,    32,  0, stream>>>(avgb, mxb, fc1, fc2, caf);
  k_mul     <<<4096, 256, 0, stream>>>(ctx, caf, f, xr, res0);
  k_conv3   <<<128,  256, 0, stream>>>(res0, c1w, c1b, res1);
  k_conv3   <<<128,  256, 0, stream>>>(res1, c2w, c2b, qsc);   // reuse dead qsc buffer
  k_final   <<<128,  256, 0, stream>>>(qsc, c3w, c3b, seg, out1, out2);
}
